// Arma_33337536151794
// MI455X (gfx1250) — matrix operand order repaired and checked by reading, not compiled, not run on hardware
//
#include <hip/hip_runtime.h>
#include <hip/hip_bf16.h>

// ---------------------------------------------------------------------------
// ARMA-GCN (2 layers, K=2 stacks) for MI455X / gfx1250.
// GEMMs use v_wmma_f32_16x16x32_bf16 (codegen-confirmed builtin).
// Edge gather/scatter uses fp32 global atomics (agg buffers fit in 192MB L2).
// ---------------------------------------------------------------------------

#define NN   100000
#define EE   3200000
#define F1   512
#define HID  16
#define NCLS 40

typedef __attribute__((ext_vector_type(16))) __bf16 v16bf;
typedef __attribute__((ext_vector_type(8)))  float  v8f;

// ---------------- edge-norm kernels ----------------------------------------

__global__ void arma_degree(const int* __restrict__ dst, float* __restrict__ deg) {
    int e = blockIdx.x * blockDim.x + threadIdx.x;
    if (e >= EE) return;
    atomicAdd(&deg[dst[e]], 1.0f);
}

__global__ void arma_dinv(float* __restrict__ deg) {
    int n = blockIdx.x * blockDim.x + threadIdx.x;
    if (n >= NN) return;
    float d = deg[n];
    deg[n] = (d > 0.0f) ? rsqrtf(d) : 0.0f;   // in-place: deg -> dinv
}

// ---------------- weight packing: bf16, column-major (col*Kdim + k) ---------

// B1: 64 cols (k0W 0-15 | k1W 16-31 | k0V 32-47 | k1V 48-63) x 512 k
__global__ void arma_pack_b1(const float* __restrict__ w1, const float* __restrict__ v1,
                             unsigned short* __restrict__ b1u) {
    int t = blockIdx.x * blockDim.x + threadIdx.x;
    if (t >= 64 * F1) return;
    int col = t >> 9, f = t & (F1 - 1);
    float val;
    if (col < 32) { int k = col >> 4, o = col & 15; val = w1[(k * F1 + f) * HID + o]; }
    else          { int c = col - 32; int k = c >> 4, o = c & 15; val = v1[(k * F1 + f) * HID + o]; }
    ((__bf16*)b1u)[t] = (__bf16)val;
}

// B2: 160 cols (k0W 0-39 | k1W 40-79 | k0V 80-119 | k1V 120-159) x 32 k (16 real + 16 zero pad)
__global__ void arma_pack_b2(const float* __restrict__ w2, const float* __restrict__ v2,
                             unsigned short* __restrict__ b2u) {
    int t = blockIdx.x * blockDim.x + threadIdx.x;
    if (t >= 160 * 32) return;
    int col = t >> 5, f = t & 31;
    float val = 0.0f;
    if (f < HID) {
        if (col < 80) { int k = col / NCLS, o = col % NCLS; val = w2[(k * HID + f) * NCLS + o]; }
        else          { int c = col - 80; int k = c / NCLS, o = c % NCLS; val = v2[(k * HID + f) * NCLS + o]; }
    }
    ((__bf16*)b2u)[t] = (__bf16)val;
}

// ---------------- layer-1 GEMM: hcat1[N x 64] = x[N x 512] @ B1 ------------
// One wave -> 16-row x 64-col strip (4 accumulators), K loop 512/32 = 16.

__global__ __launch_bounds__(256) void arma_gemm1(const float* __restrict__ x,
                                                  const unsigned short* __restrict__ b1u,
                                                  float* __restrict__ hcat1) {
    const int ntiles = NN / 16;                       // 6250
    int rt = blockIdx.x * 8 + (threadIdx.x >> 5);
    if (rt >= ntiles) return;                         // wave-uniform exit: EXEC stays all-ones
    int lane = threadIdx.x & 31;
    int g = lane >> 4, n = lane & 15;
    const __bf16* B = (const __bf16*)b1u;
    const float* xr = x + (size_t)(rt * 16 + n) * F1;

    v8f acc0 = {}, acc1 = {}, acc2 = {}, acc3 = {};
    for (int kk = 0; kk < 16; ++kk) {
        int k0 = kk * 32;
        const float4* p0 = (const float4*)(xr + k0 + g * 16);       // K = k0+g*16 .. +7
        const float4* p1 = (const float4*)(xr + k0 + g * 16 + 8);   // K = k0+g*16+8 .. +15
        float4 f0 = p0[0], f1 = p0[1], f2 = p1[0], f3 = p1[1];
        v16bf a;
        a[0]=(__bf16)f0.x; a[1]=(__bf16)f0.y; a[2]=(__bf16)f0.z; a[3]=(__bf16)f0.w;
        a[4]=(__bf16)f1.x; a[5]=(__bf16)f1.y; a[6]=(__bf16)f1.z; a[7]=(__bf16)f1.w;
        a[8]=(__bf16)f2.x; a[9]=(__bf16)f2.y; a[10]=(__bf16)f2.z; a[11]=(__bf16)f2.w;
        a[12]=(__bf16)f3.x; a[13]=(__bf16)f3.y; a[14]=(__bf16)f3.z; a[15]=(__bf16)f3.w;

        v16bf b0 = *(const v16bf*)(B + ((0 * 16 + n) * F1 + k0 + g * 16));
        v16bf b1 = *(const v16bf*)(B + ((1 * 16 + n) * F1 + k0 + g * 16));
        v16bf b2 = *(const v16bf*)(B + ((2 * 16 + n) * F1 + k0 + g * 16));
        v16bf b3 = *(const v16bf*)(B + ((3 * 16 + n) * F1 + k0 + g * 16));
        acc0 = __builtin_amdgcn_wmma_f32_16x16x32_bf16(false, a, false, b0, (short)0, acc0, false, false);
        acc1 = __builtin_amdgcn_wmma_f32_16x16x32_bf16(false, a, false, b1, (short)0, acc1, false, false);
        acc2 = __builtin_amdgcn_wmma_f32_16x16x32_bf16(false, a, false, b2, (short)0, acc2, false, false);
        acc3 = __builtin_amdgcn_wmma_f32_16x16x32_bf16(false, a, false, b3, (short)0, acc3, false, false);
    }
#pragma unroll
    for (int r = 0; r < 8; ++r) {
        float* o = hcat1 + (size_t)(rt * 16 + g * 8 + r) * 64 + n;
        o[0] = acc0[r]; o[16] = acc1[r]; o[32] = acc2[r]; o[48] = acc3[r];
    }
}

// ---------------- layer-1 scatter: agg1[dst] += norm * hcat1[src, 0:32] ----

__global__ void arma_scatter1(const int* __restrict__ src, const int* __restrict__ dst,
                              const float* __restrict__ dinv,
                              const float* __restrict__ hcat1, float* __restrict__ agg1) {
    int e = blockIdx.x * blockDim.x + threadIdx.x;
    if (e >= EE) return;
    int s = src[e], d = dst[e];
    float w = dinv[s] * dinv[d];
    const float4* hs = (const float4*)(hcat1 + (size_t)s * 64);
    float* ag = agg1 + (size_t)d * 32;
#pragma unroll
    for (int i = 0; i < 8; ++i) {
        float4 v = hs[i];
        atomicAdd(ag + i * 4 + 0, v.x * w);
        atomicAdd(ag + i * 4 + 1, v.y * w);
        atomicAdd(ag + i * 4 + 2, v.z * w);
        atomicAdd(ag + i * 4 + 3, v.w * w);
    }
}

// ---------------- layer-1 combine: relu + bias + mean over k ---------------

__global__ void arma_combine1(const float* __restrict__ agg1, const float* __restrict__ hcat1,
                              const float* __restrict__ bias1, float* __restrict__ h) {
    int t = blockIdx.x * blockDim.x + threadIdx.x;
    if (t >= NN * HID) return;
    int n = t >> 4, o = t & 15;
    float y0 = agg1[(size_t)n * 32 + o]      + hcat1[(size_t)n * 64 + 32 + o] + bias1[o];
    float y1 = agg1[(size_t)n * 32 + 16 + o] + hcat1[(size_t)n * 64 + 48 + o] + bias1[16 + o];
    h[t] = 0.5f * (fmaxf(y0, 0.0f) + fmaxf(y1, 0.0f));
}

// ---------------- layer-2 GEMM: h2cat[N x 160] = h[N x 16] @ B2 ------------
// K padded 16 -> 32 with zeros; 10 column tiles, one WMMA each.

__global__ __launch_bounds__(256) void arma_gemm2(const float* __restrict__ h,
                                                  const unsigned short* __restrict__ b2u,
                                                  float* __restrict__ h2cat) {
    const int ntiles = NN / 16;
    int rt = blockIdx.x * 8 + (threadIdx.x >> 5);
    if (rt >= ntiles) return;
    int lane = threadIdx.x & 31;
    int g = lane >> 4, n = lane & 15;
    const __bf16* B = (const __bf16*)b2u;
    const float* hr = h + (size_t)(rt * 16 + n) * HID;

    const float4* hp = (const float4*)hr;   // K = 0..15 (lane half 0)
    float4 f0 = hp[0], f1 = hp[1], f2 = hp[2], f3 = hp[3];
    v16bf a;
    a[0]=(__bf16)f0.x; a[1]=(__bf16)f0.y; a[2]=(__bf16)f0.z; a[3]=(__bf16)f0.w;
    a[4]=(__bf16)f1.x; a[5]=(__bf16)f1.y; a[6]=(__bf16)f1.z; a[7]=(__bf16)f1.w;
    a[8]=(__bf16)f2.x; a[9]=(__bf16)f2.y; a[10]=(__bf16)f2.z; a[11]=(__bf16)f2.w;
    a[12]=(__bf16)f3.x; a[13]=(__bf16)f3.y; a[14]=(__bf16)f3.z; a[15]=(__bf16)f3.w;
#pragma unroll
    for (int j = 0; j < 16; ++j) if (g) a[j] = (__bf16)0.0f; // lane half 1: K=16..31 pad

#pragma unroll
    for (int ct = 0; ct < 10; ++ct) {
        v16bf b = *(const v16bf*)(B + ((ct * 16 + n) * 32 + g * 16));
        v8f c = {};
        c = __builtin_amdgcn_wmma_f32_16x16x32_bf16(false, a, false, b, (short)0, c, false, false);
#pragma unroll
        for (int r = 0; r < 8; ++r)
            h2cat[(size_t)(rt * 16 + g * 8 + r) * 160 + ct * 16 + n] = c[r];
    }
}

// ---------------- layer-2 scatter: agg2[dst] += norm * h2cat[src, 0:80] ----
// 4 threads per edge, 20 columns each.

__global__ void arma_scatter2(const int* __restrict__ src, const int* __restrict__ dst,
                              const float* __restrict__ dinv,
                              const float* __restrict__ h2cat, float* __restrict__ agg2) {
    int t = blockIdx.x * blockDim.x + threadIdx.x;
    if (t >= EE * 4) return;
    int e = t >> 2, part = t & 3;
    int s = src[e], d = dst[e];
    float w = dinv[s] * dinv[d];
    const float4* hs = (const float4*)(h2cat + (size_t)s * 160 + part * 20);
    float* ag = agg2 + (size_t)d * 80 + part * 20;
#pragma unroll
    for (int i = 0; i < 5; ++i) {
        float4 v = hs[i];
        atomicAdd(ag + i * 4 + 0, v.x * w);
        atomicAdd(ag + i * 4 + 1, v.y * w);
        atomicAdd(ag + i * 4 + 2, v.z * w);
        atomicAdd(ag + i * 4 + 3, v.w * w);
    }
}

// ---------------- output: relu + bias + mean over k ------------------------

__global__ void arma_out(const float* __restrict__ agg2, const float* __restrict__ h2cat,
                         const float* __restrict__ bias2, float* __restrict__ out) {
    int t = blockIdx.x * blockDim.x + threadIdx.x;
    if (t >= NN * NCLS) return;
    int n = t / NCLS, o = t - n * NCLS;
    float y0 = agg2[(size_t)n * 80 + o]      + h2cat[(size_t)n * 160 + 80 + o]  + bias2[o];
    float y1 = agg2[(size_t)n * 80 + 40 + o] + h2cat[(size_t)n * 160 + 120 + o] + bias2[40 + o];
    out[t] = 0.5f * (fmaxf(y0, 0.0f) + fmaxf(y1, 0.0f));
}

// ---------------------------------------------------------------------------

static inline size_t aln(size_t x) { return (x + 511) & ~size_t(511); }

extern "C" void kernel_launch(void* const* d_in, const int* in_sizes, int n_in,
                              void* d_out, int out_size, void* d_ws, size_t ws_size,
                              hipStream_t stream) {
    const float* x   = (const float*)d_in[0];
    const int*   ei  = (const int*)d_in[1];
    const int*   src = ei;
    const int*   dst = ei + EE;
    const float* w1  = (const float*)d_in[2];
    const float* v1  = (const float*)d_in[3];
    const float* b1  = (const float*)d_in[4];
    const float* w2  = (const float*)d_in[5];
    const float* v2  = (const float*)d_in[6];
    const float* b2  = (const float*)d_in[7];
    float* out = (float*)d_out;

    // workspace layout (phase-overlapped): peak ~104 MB
    char* ws = (char*)d_ws;
    size_t off = 0;
    float*          dinv = (float*)(ws + off);           off = aln(off + (size_t)NN * 4);
    unsigned short* B1   = (unsigned short*)(ws + off);  off = aln(off + (size_t)64 * F1 * 2);
    unsigned short* B2   = (unsigned short*)(ws + off);  off = aln(off + (size_t)160 * 32 * 2);
    float*          h    = (float*)(ws + off);           off = aln(off + (size_t)NN * HID * 4);
    float*          big  = (float*)(ws + off);
    float* hcat1 = big;                        // phase A: N*64
    float* agg1  = big + (size_t)NN * 64;      // phase A: N*32
    float* h2cat = big;                        // phase B: N*160 (reuses A region)
    float* agg2  = big + (size_t)NN * 160;     // phase B: N*80

    // ---- edge norm ----
    hipMemsetAsync(dinv, 0, (size_t)NN * 4, stream);
    arma_degree<<<(EE + 255) / 256, 256, 0, stream>>>(dst, dinv);
    arma_dinv<<<(NN + 255) / 256, 256, 0, stream>>>(dinv);

    // ---- pack weights to bf16 (transposed) ----
    arma_pack_b1<<<(64 * F1 + 255) / 256, 256, 0, stream>>>(w1, v1, B1);
    arma_pack_b2<<<(160 * 32 + 255) / 256, 256, 0, stream>>>(w2, v2, B2);

    // ---- layer 1 ----
    arma_gemm1<<<(NN / 16 + 7) / 8, 256, 0, stream>>>(x, B1, hcat1);
    hipMemsetAsync(agg1, 0, (size_t)NN * 32 * 4, stream);
    arma_scatter1<<<(EE + 255) / 256, 256, 0, stream>>>(src, dst, dinv, hcat1, agg1);
    arma_combine1<<<(NN * HID + 255) / 256, 256, 0, stream>>>(agg1, hcat1, b1, h);

    // ---- layer 2 ----
    arma_gemm2<<<(NN / 16 + 7) / 8, 256, 0, stream>>>(h, B2, h2cat);
    hipMemsetAsync(agg2, 0, (size_t)NN * 80 * 4, stream);
    arma_scatter2<<<(EE * 4 + 255) / 256, 256, 0, stream>>>(src, dst, dinv, h2cat, agg2);
    arma_out<<<(NN * NCLS + 255) / 256, 256, 0, stream>>>(agg2, h2cat, b2, out);
}
